// DisenLayer_9174050144868
// MI455X (gfx1250) — matrix operand order repaired and checked by reading, not compiled, not run on hardware
//
#include <hip/hip_runtime.h>

#define KF 3
#define DD 128
#define KD (KF*DD)   // 384

typedef __bf16 bf16_t;
typedef __attribute__((ext_vector_type(16))) __bf16 v16bf;
typedef __attribute__((ext_vector_type(8)))  float  v8f;

// order-preserving float <-> uint key (for atomicMax on possibly-negative logits)
__device__ __forceinline__ unsigned fenc(float f){
  unsigned u = __float_as_uint(f);
  return (u >> 31) ? ~u : (u | 0x80000000u);
}
__device__ __forceinline__ float fdec(unsigned k){
  unsigned u = (k >> 31) ? (k ^ 0x80000000u) : ~k;
  return __uint_as_float(u);
}

__global__ void k_zero32(unsigned* __restrict__ p, size_t n){
  size_t i = (size_t)blockIdx.x*blockDim.x + threadIdx.x;
  if (i < n) p[i] = 0u;
}

__global__ void k_hist(const int* __restrict__ et, int E, int Etot, int loopType,
                       unsigned* __restrict__ counts){
  int e = blockIdx.x*blockDim.x + threadIdx.x;
  if (e >= Etot) return;
  int t = (e < E) ? et[e] : loopType;
  atomicAdd(&counts[t], 1u);
}

__global__ void k_scan(const unsigned* __restrict__ counts, unsigned* __restrict__ off,
                       unsigned* __restrict__ toff, unsigned* __restrict__ numTiles, int ntypes){
  if (threadIdx.x == 0 && blockIdx.x == 0){
    unsigned o = 0, to = 0;
    for (int t = 0; t < ntypes; ++t){
      off[t] = o; toff[t] = to;
      o  += counts[t];
      to += (counts[t] + 15u) >> 4;
    }
    off[ntypes] = o; toff[ntypes] = to;
    *numTiles = to;
  }
}

__global__ void k_desc(const unsigned* __restrict__ counts, const unsigned* __restrict__ off,
                       const unsigned* __restrict__ toff, unsigned* __restrict__ desc){
  int t = blockIdx.x;
  unsigned nT = (counts[t] + 15u) >> 4;
  unsigned base = toff[t];
  for (unsigned q = threadIdx.x; q < nT; q += blockDim.x){
    desc[2*(base+q)+0] = (unsigned)t;
    desc[2*(base+q)+1] = off[t] + q*16u;   // absolute start in sorted order
  }
}

__global__ void k_scatter(const int* __restrict__ et, int E, int Etot, int loopType,
                          const unsigned* __restrict__ off, unsigned* __restrict__ cursor,
                          unsigned* __restrict__ sorted){
  int e = blockIdx.x*blockDim.x + threadIdx.x;
  if (e >= Etot) return;
  int t = (e < E) ? et[e] : loopType;
  unsigned p = atomicAdd(&cursor[t], 1u);
  sorted[off[t]+p] = (unsigned)e;
}

// g[t][k][m] = sum_n r_t[(m+n) mod 128] * att[k][n]
// (folds the circulant * attention dot so pass-1 logits need no GEMM)
__global__ __launch_bounds__(DD)
void k_gatt(const float* __restrict__ rel_embed, const float* __restrict__ loop_rel,
            const float* __restrict__ att, float* __restrict__ g, int nrel2){
  __shared__ float r2[2*DD];
  __shared__ float attS[KF][DD];
  const int t = blockIdx.x, tid = threadIdx.x;
  const float* rp = (t < nrel2) ? (rel_embed + (size_t)t*DD) : loop_rel;
  { float v = rp[tid]; r2[tid] = v; r2[DD + tid] = v; }
  for (int idx = tid; idx < KF*DD; idx += blockDim.x)
    attS[idx/DD][idx%DD] = att[idx];
  __syncthreads();
  const int m = tid;
  for (int k = 0; k < KF; ++k){
    float acc = 0.f;
    #pragma unroll 8
    for (int n = 0; n < DD; ++n) acc += r2[m + n] * attS[k][n];
    g[(size_t)t*KD + k*DD + m] = acc;
  }
}

// pass 1: one wave per sorted edge: logit[e,k] = leakyrelu( sum_m x[j,k,m]*w_t[k,m]*g_t[k,m] )
__global__ __launch_bounds__(256)
void k_logit(const float* __restrict__ x, const float* __restrict__ rel_weight,
             const float* __restrict__ g, const int* __restrict__ edge_index,
             const int* __restrict__ edge_type, const unsigned* __restrict__ sorted,
             unsigned* __restrict__ iidx, float* __restrict__ logits,
             unsigned* __restrict__ gmax, int E, int Etot, int nrel2){
  int wid  = (int)((blockIdx.x*(unsigned)blockDim.x + threadIdx.x) >> 5);
  int lane = threadIdx.x & 31;
  if (wid >= Etot) return;
  int e = (int)sorted[wid];
  int i_, j_, t;
  if (e < E){ i_ = edge_index[e]; j_ = edge_index[E + e]; t = edge_type[e]; }
  else      { i_ = e - E; j_ = i_; t = nrel2; }
  if (lane == 0) iidx[wid] = (unsigned)i_;
  const float* xr = x          + (size_t)j_*KD;
  const float* wr = rel_weight + (size_t)t*KD;
  const float* gr = g          + (size_t)t*KD;
  for (int k = 0; k < KF; ++k){
    float p = 0.f;
    #pragma unroll
    for (int mm = lane; mm < DD; mm += 32){
      int c = k*DD + mm;
      p += xr[c] * wr[c] * gr[c];
    }
    p += __shfl_xor(p, 1, 32);
    p += __shfl_xor(p, 2, 32);
    p += __shfl_xor(p, 4, 32);
    p += __shfl_xor(p, 8, 32);
    p += __shfl_xor(p, 16, 32);
    if (lane == 0){
      float l = (p > 0.f) ? p : 0.2f*p;             // leaky_relu(0.2)
      logits[(size_t)wid*KF + k] = l;
      atomicMax(&gmax[(size_t)i_*KF + k], fenc(l));
    }
  }
}

__global__ void k_sumexp(const float* __restrict__ logits, const unsigned* __restrict__ iidx,
                         const unsigned* __restrict__ gmax, float* __restrict__ gsum, int Etot){
  int idx = blockIdx.x*blockDim.x + threadIdx.x;
  if (idx >= Etot*KF) return;
  int s = idx / KF, k = idx % KF;
  int i = (int)iidx[s];
  float mx = fdec(gmax[(size_t)i*KF + k]);
  atomicAdd(&gsum[(size_t)i*KF + k], __expf(logits[idx] - mx));
}

// ---------------------------------------------------------------------------
// Single WMMA GEMM pass: per 16-edge tile of one relation type,
//   msg[e,k,n] = sum_m (alpha[e,k]*x[j,k,m]*w_t[k,m]) * r_t[(m+n) mod 128]
// alpha folded into A; epilogue is a pure atomic scatter-add with
// precomputed u32 row offsets and a uniform full-tile fast path.
// ---------------------------------------------------------------------------
__global__ __launch_bounds__(128)
void k_gemm2(const float* __restrict__ x,
             const float* __restrict__ rel_embed,
             const float* __restrict__ loop_rel,
             const float* __restrict__ rel_weight,
             const int*   __restrict__ edge_index,
             const unsigned* __restrict__ sorted,
             const unsigned* __restrict__ off,
             const unsigned* __restrict__ desc,
             const unsigned* __restrict__ numTilesP,
             const float* __restrict__ logits,
             const unsigned* __restrict__ gmax,
             const float* __restrict__ gsum,
             float* __restrict__ outAcc,
             int E, int nrel2)
{
  __shared__ bf16_t r2[2*DD];           // r_t duplicated: circulant source
  __shared__ bf16_t aT[KF][16][DD];     // A = alpha * x[j] * w_t  (bf16)
  __shared__ int    iS[16], jS[16];
  __shared__ float  al[16][KF];

  unsigned nt = *numTilesP;
  if (blockIdx.x >= nt) return;
  const unsigned t      = desc[2*blockIdx.x];
  const unsigned start  = desc[2*blockIdx.x + 1];
  const unsigned end    = off[t+1];
  const unsigned nvalid = end - start;      // rows in this tile (1..16)
  const int tid = threadIdx.x;

  if (tid < DD){
    const float* rp = (t < (unsigned)nrel2) ? (rel_embed + (size_t)t*DD) : loop_rel;
    bf16_t v = (bf16_t)rp[tid];
    r2[tid] = v; r2[DD + tid] = v;      // duplicated -> index (m+n) without mod
  }
  if (tid < 16){
    unsigned s = start + (unsigned)tid;
    int valid = ((unsigned)tid < nvalid) ? 1 : 0;
    int e = valid ? (int)sorted[s] : 0;
    int i_, j_;
    if (e < E){ i_ = edge_index[e]; j_ = edge_index[E + e]; }   // head, tail
    else      { i_ = e - E; j_ = i_; }                          // self loop
    iS[tid] = i_; jS[tid] = j_;
    #pragma unroll
    for (int k = 0; k < KF; ++k){
      float a = 0.f;
      if (valid){
        float l  = logits[(size_t)s*KF + k];
        float mx = fdec(gmax[(size_t)i_*KF + k]);
        float sm = gsum[(size_t)i_*KF + k];
        a = __expf(l - mx) / (sm + 1e-16f);         // softmax weight
      }
      al[tid][k] = a;
    }
  }
  __syncthreads();

  const float* wt = rel_weight + (size_t)t*KD;
  for (int idx = tid; idx < 16*KD; idx += blockDim.x){
    int row = idx / KD, c = idx % KD;
    float v = ((unsigned)row < nvalid)
            ? x[(size_t)jS[row]*KD + c] * wt[c] * al[row][c/DD] : 0.0f;
    aT[c/DD][row][c%DD] = (bf16_t)v;
  }
  __syncthreads();

  const int wave  = tid >> 5;
  const int lane  = tid & 31;
  const int mrow  = lane & 15;               // A row / B,C column within tile
  const int hiA   = (lane >= 16) ? 8  : 0;   // A: K-halves split across lanes
  const int koffB = (lane >= 16) ? 16 : 0;   // B: K 0-15 vs 16-31
  const int hiR   = (lane >= 16) ? 8  : 0;   // C rows r / 8+r

  // per-lane destination row offsets (u32 element offsets; hoisted out of loops)
  unsigned rowOff[8];
  #pragma unroll
  for (int r = 0; r < 8; ++r)
    rowOff[r] = (unsigned)iS[r + hiR] * (unsigned)KD;
  const bool full = (nvalid >= 16u);         // block-uniform fast path

  for (int k = 0; k < KF; ++k){
    v16bf afr[4];
    #pragma unroll
    for (int kc = 0; kc < 4; ++kc){
      #pragma unroll
      for (int i2 = 0; i2 < 16; ++i2){
        int kk = kc*32 + i2 + ((i2 >= 8) ? 8 : 0) + hiA;   // ISA 16-bit A layout
        afr[kc][i2] = aT[k][mrow][kk];
      }
    }
    #pragma unroll
    for (int q = 0; q < 2; ++q){
      const int nc = wave*2 + q;             // 4 waves x 2 = 8 column chunks
      v8f c = {0.f,0.f,0.f,0.f,0.f,0.f,0.f,0.f};
      #pragma unroll
      for (int kc = 0; kc < 4; ++kc){
        v16bf b;
        #pragma unroll
        for (int i2 = 0; i2 < 16; ++i2)      // circulant: B[kk][n] = r[(kk+n)%128]
          b[i2] = r2[kc*32 + i2 + ((i2 >= 8) ? 8 : 0) + hiA + nc*16 + mrow];
        c = __builtin_amdgcn_wmma_f32_16x16x32_bf16(false, afr[kc], false, b,
                                                    (short)0, c, false, false);
      }
      const unsigned colOff = (unsigned)(k*DD + nc*16 + mrow);
      if (full){
        #pragma unroll
        for (int r = 0; r < 8; ++r)
          atomicAdd(&outAcc[rowOff[r] + colOff], c[r]);
      } else {
        #pragma unroll
        for (int r = 0; r < 8; ++r)
          if ((unsigned)(r + hiR) < nvalid)
            atomicAdd(&outAcc[rowOff[r] + colOff], c[r]);
      }
    }
  }
}

__global__ __launch_bounds__(KD)
void k_bnstats(const float* __restrict__ h, int N, float* __restrict__ bsum,
               float* __restrict__ bsq){
  int f  = threadIdx.x;                 // 384 features, coalesced across threads
  int r0 = blockIdx.x * 256;
  int r1 = min(r0 + 256, N);
  float s = 0.f, sq = 0.f;
  for (int r = r0; r < r1; ++r){
    float v = h[(size_t)r*KD + f];
    s += v; sq += v*v;
  }
  atomicAdd(&bsum[f], s);
  atomicAdd(&bsq[f], sq);
}

__global__ void k_bnapply(float* __restrict__ h, const float* __restrict__ bsum,
                          const float* __restrict__ bsq, const float* __restrict__ gamma,
                          const float* __restrict__ beta, int N){
  size_t idx = (size_t)blockIdx.x*blockDim.x + threadIdx.x;
  if (idx >= (size_t)N*KD) return;
  int f = (int)(idx % KD);
  float inv = 1.0f / (float)N;
  float mu  = bsum[f]*inv;
  float var = bsq[f]*inv - mu*mu;
  h[idx] = (h[idx] - mu) * rsqrtf(var + 1e-5f) * gamma[f] + beta[f];
}

__global__ void k_relout(const float* __restrict__ rel_embed, const float* __restrict__ w_rel,
                         float* __restrict__ out, int nrel2){
  int idx = blockIdx.x*blockDim.x + threadIdx.x;
  if (idx >= nrel2*DD) return;
  int t = idx / DD, n = idx % DD;
  const float* r = rel_embed + (size_t)t*DD;
  float acc = 0.f;
  #pragma unroll 8
  for (int m = 0; m < DD; ++m) acc += r[m] * w_rel[m*DD + n];
  out[idx] = acc;
}

extern "C" void kernel_launch(void* const* d_in, const int* in_sizes, int n_in,
                              void* d_out, int out_size, void* d_ws, size_t ws_size,
                              hipStream_t stream){
  const float* x          = (const float*)d_in[0];
  const float* rel_embed  = (const float*)d_in[1];
  const float* loop_rel   = (const float*)d_in[2];
  const float* rel_weight = (const float*)d_in[3];
  const float* att_weight = (const float*)d_in[4];
  const float* w_rel      = (const float*)d_in[5];
  const float* bn_gamma   = (const float*)d_in[6];
  const float* bn_beta    = (const float*)d_in[7];
  const int*   edge_index = (const int*)d_in[8];
  const int*   edge_type  = (const int*)d_in[9];

  const int E      = in_sizes[8] / 2;
  const int N      = in_sizes[0] / KD;
  const int nrel2  = in_sizes[1] / DD;      // 2*N_REL = 400 (loop type == nrel2)
  const int ntypes = nrel2 + 1;
  const int Etot   = E + N;
  const int NT_MAX = (Etot + 15)/16 + ntypes;

  char* w = (char*)d_ws;
  auto carve = [&](size_t bytes)->char*{
    char* p = w; w += (bytes + 255) & ~(size_t)255; return p;
  };
  unsigned* counts   = (unsigned*)carve((size_t)ntypes*4);
  unsigned* off      = (unsigned*)carve((size_t)(ntypes+1)*4);
  unsigned* toff     = (unsigned*)carve((size_t)(ntypes+1)*4);
  unsigned* cursor   = (unsigned*)carve((size_t)ntypes*4);
  unsigned* numTiles = (unsigned*)carve(4);
  unsigned* sorted   = (unsigned*)carve((size_t)Etot*4);
  unsigned* iidx     = (unsigned*)carve((size_t)Etot*4);
  unsigned* desc     = (unsigned*)carve((size_t)NT_MAX*8);
  float*    logits   = (float*)carve((size_t)Etot*KF*4);
  unsigned* gmax     = (unsigned*)carve((size_t)N*KF*4);
  float*    gsum     = (float*)carve((size_t)N*KF*4);
  float*    gbuf     = (float*)carve((size_t)ntypes*KD*4);
  float*    bsum     = (float*)carve(KD*4);
  float*    bsq      = (float*)carve(KD*4);
  (void)ws_size; (void)n_in; (void)out_size;

  float* entity = (float*)d_out;               // [N, K, D]
  float* relout = entity + (size_t)N*KD;       // [nrel2, D]

  auto Z = [&](void* p, size_t n32){
    k_zero32<<<(unsigned)((n32 + 255)/256), 256, 0, stream>>>((unsigned*)p, n32);
  };
  Z(counts, (size_t)ntypes);  Z(cursor, (size_t)ntypes);
  Z(gmax, (size_t)N*KF);      Z(gsum, (size_t)N*KF);
  Z(bsum, KD);                Z(bsq, KD);
  Z(entity, (size_t)N*KD);

  const int B = 256;
  k_hist<<<(Etot+B-1)/B, B, 0, stream>>>(edge_type, E, Etot, nrel2, counts);
  k_scan<<<1, 32, 0, stream>>>(counts, off, toff, numTiles, ntypes);
  k_desc<<<ntypes, 64, 0, stream>>>(counts, off, toff, desc);
  k_scatter<<<(Etot+B-1)/B, B, 0, stream>>>(edge_type, E, Etot, nrel2, off, cursor, sorted);

  k_gatt<<<ntypes, DD, 0, stream>>>(rel_embed, loop_rel, att_weight, gbuf, nrel2);
  {
    unsigned thr = 256, waves_per_blk = thr/32;
    unsigned blocks = (unsigned)((Etot + waves_per_blk - 1) / waves_per_blk);
    k_logit<<<blocks, thr, 0, stream>>>(x, rel_weight, gbuf, edge_index, edge_type,
        sorted, iidx, logits, gmax, E, Etot, nrel2);
  }
  k_sumexp<<<(Etot*KF + B - 1)/B, B, 0, stream>>>(logits, iidx, gmax, gsum, Etot);

  k_gemm2<<<NT_MAX, 128, 0, stream>>>(x, rel_embed, loop_rel, rel_weight,
      edge_index, sorted, off, desc, numTiles, logits, gmax, gsum, entity, E, nrel2);

  k_bnstats<<<(N + 255)/256, KD, 0, stream>>>(entity, N, bsum, bsq);
  k_bnapply<<<(unsigned)(((size_t)N*KD + B - 1)/B), B, 0, stream>>>(entity, bsum, bsq,
      bn_gamma, bn_beta, N);
  k_relout<<<(nrel2*DD + B - 1)/B, B, 0, stream>>>(rel_embed, w_rel, relout, nrel2);
}
